// Net_8100308320579
// MI455X (gfx1250) — compile-verified, matrix operand order checked by reading, not run on hardware
//
#include <hip/hip_runtime.h>
#include <hip/hip_bf16.h>

#define NN 50000
#define EE 800000
#define FIN 128
#define DIMF 256
#define NCLS 10
#define NGRAPH 256

typedef __attribute__((ext_vector_type(16))) __bf16          v16bf;
typedef __attribute__((ext_vector_type(16))) unsigned short  v16u;
typedef __attribute__((ext_vector_type(8)))  unsigned short  v8u;
typedef __attribute__((ext_vector_type(8)))  float           v8f;

// ---------- helpers ----------
__device__ __forceinline__ unsigned short f2bf(float f) {
    unsigned u = __builtin_bit_cast(unsigned, f);
    unsigned r = (u >> 16) & 1u;
    u += 0x7fffu + r;                       // round-to-nearest-even
    return (unsigned short)(u >> 16);
}

// Build A fragment from lane-local pointer (ISA 16-bit A layout):
// lane half 0: K groups {0..7, 16..23}; half 1: {8..15, 24..31} (ap pre-offset by half*8)
__device__ __forceinline__ v16bf make_a(const unsigned short* ap) {
    v8u lo = *(const v8u*)ap;          // 16B
    v8u hi = *(const v8u*)(ap + 16);   // 16B
    v16u r;
#pragma unroll
    for (int i = 0; i < 8; ++i) { r[i] = lo[i]; r[i + 8] = hi[i]; }
    return __builtin_bit_cast(v16bf, r);
}

__device__ __forceinline__ v16bf ld_b(const unsigned short* p) {
    return __builtin_bit_cast(v16bf, *(const v16u*)p);
}

// Accumulate `ksteps` k-steps (32 each) of one A matrix against 4 adjacent B tiles.
// Fixed A base/lda per span -> inner loop is pure pointer increments.
__device__ __forceinline__ void gemm_span(const unsigned short* __restrict__ A, int lda,
                                          int row0, int ksteps,
                                          const unsigned short* __restrict__ bp,
                                          size_t bstep, int lane,
                                          v8f& acc0, v8f& acc1, v8f& acc2, v8f& acc3) {
    const unsigned short* ap =
        A + (size_t)(row0 + (lane & 15)) * lda + ((lane >> 4) & 1) * 8;
#pragma unroll 2
    for (int s = 0; s < ksteps; ++s) {
        v16bf a  = make_a(ap);
        v16bf b0 = ld_b(bp + 0 * 512);   // 4 N-tiles: constant 1KB-apart offsets
        v16bf b1 = ld_b(bp + 1 * 512);
        v16bf b2 = ld_b(bp + 2 * 512);
        v16bf b3 = ld_b(bp + 3 * 512);
        acc0 = __builtin_amdgcn_wmma_f32_16x16x32_bf16(false, a, false, b0, (short)0, acc0, false, false);
        acc1 = __builtin_amdgcn_wmma_f32_16x16x32_bf16(false, a, false, b1, (short)0, acc1, false, false);
        acc2 = __builtin_amdgcn_wmma_f32_16x16x32_bf16(false, a, false, b2, (short)0, acc2, false, false);
        acc3 = __builtin_amdgcn_wmma_f32_16x16x32_bf16(false, a, false, b3, (short)0, acc3, false, false);
        ap += 32;
        bp += bstep;
    }
}

// ---------- utility kernels ----------
__global__ void k_zero(float4* p, long long n4) {
    long long t = (long long)blockIdx.x * blockDim.x + threadIdx.x;
    long long s = (long long)gridDim.x * blockDim.x;
    float4 z = {0.f, 0.f, 0.f, 0.f};
    for (; t < n4; t += s) p[t] = z;
}

__global__ void k_f2bf(const float* src, unsigned short* dst, long long n) {
    long long t = (long long)blockIdx.x * blockDim.x + threadIdx.x;
    long long s = (long long)gridDim.x * blockDim.x;
    for (; t < n; t += s) dst[t] = f2bf(src[t]);
}

// scatter-add edge messages (4 features per thread): agg[dst] += h[src] * ew
__global__ void k_edge_scatter(const float* __restrict__ h,
                               const int* __restrict__ src, const int* __restrict__ dst,
                               const float* __restrict__ ew, float* __restrict__ agg,
                               long long total4, int kshift) {
    long long t = (long long)blockIdx.x * blockDim.x + threadIdx.x;
    long long s = (long long)gridDim.x * blockDim.x;
    const int qshift = kshift - 2;
    const int qmask  = (1 << qshift) - 1;
    for (; t < total4; t += s) {
        int e = (int)(t >> qshift);
        int q = (int)(t & qmask);
        float w = ew[e];
        const float4 xv = ((const float4*)(h + ((size_t)src[e] << kshift)))[q];
        float* ap = agg + ((size_t)dst[e] << kshift) + q * 4;
        unsafeAtomicAdd(ap + 0, xv.x * w);
        unsafeAtomicAdd(ap + 1, xv.y * w);
        unsafeAtomicAdd(ap + 2, xv.z * w);
        unsafeAtomicAdd(ap + 3, xv.w * w);
    }
}

// pooled[batch[n]] += h[n]   (256 features, 4 per thread)
__global__ void k_pool(const float* __restrict__ h, const int* __restrict__ batch,
                       float* __restrict__ pooled, long long total4) {
    long long t = (long long)blockIdx.x * blockDim.x + threadIdx.x;
    long long s = (long long)gridDim.x * blockDim.x;
    for (; t < total4; t += s) {
        int n = (int)(t >> 6);
        int q = (int)(t & 63);
        const float4 xv = ((const float4*)(h + ((size_t)n << 8)))[q];
        float* ap = pooled + ((size_t)batch[n] << 8) + q * 4;
        unsafeAtomicAdd(ap + 0, xv.x);
        unsafeAtomicAdd(ap + 1, xv.y);
        unsafeAtomicAdd(ap + 2, xv.z);
        unsafeAtomicAdd(ap + 3, xv.w);
    }
}

// pack concat weight [W0 ; W1] (f32, row-major [K,256]) into WMMA B-fragment layout (bf16)
__global__ void k_pack_b(const float* __restrict__ W0, const float* __restrict__ W1,
                         int K0, int Ktot, unsigned short* __restrict__ Bp) {
    const int tiles_n = DIMF / 16;
    long long total = (long long)(Ktot / 32) * tiles_n * 32;
    long long t = (long long)blockIdx.x * blockDim.x + threadIdx.x;
    long long s = (long long)gridDim.x * blockDim.x;
    for (; t < total; t += s) {
        int kt   = (int)(t / (tiles_n * 32));
        int rem  = (int)(t % (tiles_n * 32));
        int nt   = rem / 32;
        int lane = rem & 31;
        int half = (lane >> 4) & 1;
        int n    = nt * 16 + (lane & 15);
#pragma unroll
        for (int i = 0; i < 16; ++i) {
            int k = kt * 32 + ((i >> 3) * 16) + half * 8 + (i & 7);
            const float* W = (k < K0) ? W0 : W1;
            int kk = (k < K0) ? k : (k - K0);
            Bp[(size_t)t * 16 + i] = f2bf(W[(size_t)kk * DIMF + n]);
        }
    }
}

// C[M,256] = act( [A0|A1] @ Bp + bias ).
// Each wave: one 16-row M-tile x four 16-col N-tiles; 8 waves/block stack in M.
// Optionally also emits bf16 copy of C (Cb) for the next layer's A-operand.
__global__ __launch_bounds__(256) void k_gemm_wmma(
    const unsigned short* __restrict__ A0, const unsigned short* __restrict__ A1,
    int K0, int Ktot, const unsigned short* __restrict__ Bp,
    const float* __restrict__ bias, float* __restrict__ C,
    unsigned short* __restrict__ Cb, int tiles_m, int relu) {
    const int tiles_n = DIMF / 16;
    int lane = threadIdx.x & 31;
    int wave = threadIdx.x >> 5;
    int mt = blockIdx.x * 8 + wave;          // wave-uniform
    if (mt >= tiles_m) return;               // whole-wave exit: EXEC all-1s for WMMA
    int ng = blockIdx.y;                     // group of 4 N-tiles

    v8f acc0 = {}, acc1 = {}, acc2 = {}, acc3 = {};
    const size_t bstep = (size_t)tiles_n * 512;            // ushorts per k-step
    const unsigned short* bp0 =
        Bp + ((size_t)(ng * 4) * 32 + lane) * 16;          // kt = 0
    gemm_span(A0, K0, mt * 16, K0 / 32, bp0, bstep, lane, acc0, acc1, acc2, acc3);
    if (Ktot > K0) {
        const unsigned short* bp1 =
            Bp + ((size_t)((K0 / 32) * tiles_n + ng * 4) * 32 + lane) * 16;
        gemm_span(A1, Ktot - K0, mt * 16, (Ktot - K0) / 32, bp1, bstep, lane,
                  acc0, acc1, acc2, acc3);
    }

    int half = (lane >> 4) & 1;
    int nl   = lane & 15;
#pragma unroll
    for (int j = 0; j < 4; ++j) {
        const v8f& acc = (j == 0) ? acc0 : (j == 1) ? acc1 : (j == 2) ? acc2 : acc3;
        int n  = (ng * 4 + j) * 16 + nl;
        float bb = bias[n];
#pragma unroll
        for (int r = 0; r < 8; ++r) {
            int m = mt * 16 + r + half * 8;
            float v = acc[r] + bb;
            if (relu) v = v > 0.f ? v : 0.f;
            C[(size_t)m * DIMF + n] = v;
            if (Cb) Cb[(size_t)m * DIMF + n] = f2bf(v);
        }
    }
}

// fc2 + log_softmax: one wave per graph
__global__ __launch_bounds__(32) void k_fc2_logsoftmax(
    const float* __restrict__ h, const float* __restrict__ W,
    const float* __restrict__ b, float* __restrict__ out) {
    int g = blockIdx.x;
    int lane = threadIdx.x;
    float logit = -INFINITY;
    if (lane < NCLS) {
        float s = b[lane];
        const float* hr = h + (size_t)g * DIMF;
#pragma unroll 4
        for (int k = 0; k < DIMF; ++k) s += hr[k] * W[k * NCLS + lane];
        logit = s;
    }
    float m = logit;
    for (int off = 16; off; off >>= 1) m = fmaxf(m, __shfl_xor(m, off, 32));
    float e = (lane < NCLS) ? __expf(logit - m) : 0.f;
    float se = e;
    for (int off = 16; off; off >>= 1) se += __shfl_xor(se, off, 32);
    if (lane < NCLS) out[(size_t)g * NCLS + lane] = logit - m - logf(se);
}

// ---------- orchestration ----------
extern "C" void kernel_launch(void* const* d_in, const int* in_sizes, int n_in,
                              void* d_out, int out_size, void* d_ws, size_t ws_size,
                              hipStream_t stream) {
    (void)in_sizes; (void)n_in; (void)out_size; (void)ws_size;

    const float* x     = (const float*)d_in[0];
    const int*   ei    = (const int*)d_in[1];
    const int*   batch = (const int*)d_in[2];
    const float* ew    = (const float*)d_in[3];
    const float* Wrel[5]  = {(const float*)d_in[4],  (const float*)d_in[7],
                             (const float*)d_in[10], (const float*)d_in[13],
                             (const float*)d_in[16]};
    const float* brel[5]  = {(const float*)d_in[5],  (const float*)d_in[8],
                             (const float*)d_in[11], (const float*)d_in[14],
                             (const float*)d_in[17]};
    const float* Wroot[5] = {(const float*)d_in[6],  (const float*)d_in[9],
                             (const float*)d_in[12], (const float*)d_in[15],
                             (const float*)d_in[18]};
    const float* Wfc1 = (const float*)d_in[19];
    const float* bfc1 = (const float*)d_in[20];
    const float* Wfc2 = (const float*)d_in[21];
    const float* bfc2 = (const float*)d_in[22];
    float* out = (float*)d_out;

    const int* src = ei;
    const int* dst = ei + EE;

    // bump-allocate workspace
    size_t off = 0;
    auto alloc = [&](size_t bytes) -> char* {
        char* p = (char*)d_ws + off;
        off = (off + bytes + 255) & ~(size_t)255;
        return p;
    };
    float*          agg     = (float*)alloc((size_t)NN * DIMF * 4);
    float*          h       = (float*)alloc((size_t)NN * DIMF * 4);
    unsigned short* aggb    = (unsigned short*)alloc((size_t)NN * DIMF * 2);
    unsigned short* hbA     = (unsigned short*)alloc((size_t)NN * DIMF * 2);
    unsigned short* hbB     = (unsigned short*)alloc((size_t)NN * DIMF * 2);
    unsigned short* Bp      = (unsigned short*)alloc((size_t)2 * DIMF * DIMF * 2);
    float*          pooled  = (float*)alloc((size_t)NGRAPH * DIMF * 4);
    unsigned short* pooledb = (unsigned short*)alloc((size_t)NGRAPH * DIMF * 2);
    float*          fc1o    = (float*)alloc((size_t)NGRAPH * DIMF * 4);

    const int TB = 256;
    const int GS = 2048;

    unsigned short* hb_in  = hbA;   // bf16 activations for GEMM A-operand
    unsigned short* hb_out = hbB;

    // layer 0 A-operand: convert x once
    k_f2bf<<<GS, TB, 0, stream>>>(x, hb_in, (long long)NN * FIN);

    for (int layer = 0; layer < 5; ++layer) {
        const int K = (layer == 0) ? FIN : DIMF;
        const int kshift = (layer == 0) ? 7 : 8;
        const float* hin = (layer == 0) ? x : h;
        const long long nk = (long long)NN * K;

        k_zero<<<GS, TB, 0, stream>>>((float4*)agg, nk >> 2);
        k_edge_scatter<<<4096, TB, 0, stream>>>(hin, src, dst, ew, agg,
                                                (long long)EE << (kshift - 2), kshift);
        k_f2bf<<<GS, TB, 0, stream>>>(agg, aggb, nk);
        k_pack_b<<<64, TB, 0, stream>>>(Wrel[layer], Wroot[layer], K, 2 * K, Bp);

        // GEMM writes f32 h (for next scatter) and bf16 hb_out (for next GEMM A)
        dim3 grid((NN / 16 + 7) / 8, DIMF / 16 / 4);
        k_gemm_wmma<<<grid, TB, 0, stream>>>(aggb, hb_in, K, 2 * K, Bp,
                                             brel[layer], h, hb_out, NN / 16, 1);
        unsigned short* t = hb_in; hb_in = hb_out; hb_out = t;
    }

    // global pooling by graph id
    k_zero<<<GS, TB, 0, stream>>>((float4*)pooled, (long long)(NGRAPH * DIMF) >> 2);
    k_pool<<<GS, TB, 0, stream>>>(h, batch, pooled, ((long long)NN * DIMF) >> 2);

    // fc1 (WMMA) + relu
    k_f2bf<<<64, TB, 0, stream>>>(pooled, pooledb, (long long)NGRAPH * DIMF);
    k_pack_b<<<32, TB, 0, stream>>>(Wfc1, Wfc1, DIMF, DIMF, Bp);
    dim3 g1((NGRAPH / 16 + 7) / 8, DIMF / 16 / 4);
    k_gemm_wmma<<<g1, TB, 0, stream>>>(pooledb, pooledb, DIMF, DIMF, Bp,
                                       bfc1, fc1o, (unsigned short*)nullptr,
                                       NGRAPH / 16, 1);

    // fc2 + log_softmax
    k_fc2_logsoftmax<<<NGRAPH, 32, 0, stream>>>(fc1o, Wfc2, bfc2, out);
}
